// NPCSAGE_26388279067151
// MI455X (gfx1250) — matrix operand order repaired and checked by reading, not compiled, not run on hardware
//
#include <hip/hip_runtime.h>

typedef __attribute__((ext_vector_type(16))) __bf16 v16bf;
typedef __attribute__((ext_vector_type(8)))  float  v8f;

__device__ __forceinline__ unsigned short f2bf(float x) {
    union { float f; unsigned u; } v; v.f = x;
    unsigned r = v.u + 0x7FFFu + ((v.u >> 16) & 1u);   // RNE
    return (unsigned short)(r >> 16);
}

__device__ __forceinline__ float relu_maybe(float x, int r) {
    return r ? fmaxf(x, 0.f) : x;
}

// ---------------------------------------------------------------------------
// Pack [W_self ; W_neigh] (each Kin x Nout, f32 row-major) into bf16 WMMA
// B-fragment order: half index = ((nt*NKT + kt)*32 + lane)*16 + e
// where N = nt*16 + (lane&15), K' = kt*32 + (lane>>4)*16 + e.
// Columns >= Nout padded with zero (layer 2: N=47 -> Npad=48).
// ---------------------------------------------------------------------------
__global__ void sage_pack_w(const float* __restrict__ Ws,
                            const float* __restrict__ Wn,
                            unsigned short* __restrict__ Wp,
                            int Kin, int Nout, int NKT, int NTP) {
    int id = blockIdx.x * 256 + threadIdx.x;
    int total = NTP * NKT * 512;
    if (id >= total) return;
    int e    = id & 15;
    int t    = id >> 4;
    int lane = t & 31;  t >>= 5;
    int kt   = t % NKT;
    int nt   = t / NKT;
    int n    = nt * 16 + (lane & 15);
    int kp   = kt * 32 + (lane >> 4) * 16 + e;
    float v = 0.f;
    if (n < Nout)
        v = (kp < Kin) ? Ws[(size_t)kp * Nout + n] : Wn[(size_t)(kp - Kin) * Nout + n];
    Wp[id] = f2bf(v);
}

// ---------------------------------------------------------------------------
// Fused SAGE layer: gather+mean(+perm,+relu_in) -> LDS bf16 A panel -> WMMA
// block = 256 threads (8 wave32), each block = 16 dst rows, waves split N-tiles
// ---------------------------------------------------------------------------
__global__ void __launch_bounds__(256)
sage_layer(const float* __restrict__ in,       // [nsrc x Kin] feature table
           const int*   __restrict__ perm,     // nullable row permutation
           const int*   __restrict__ idx,      // [ndst x fan]
           const unsigned short* __restrict__ Wp, // packed bf16 [NTP][NKT][32][16]
           const float* __restrict__ bias,     // [Nout]
           float*       __restrict__ out,      // [ndst x Nout]
           int Kin, int fan, int Nout, int Npad,
           int relu_in, int relu_out) {
    __shared__ unsigned short Alds[16 * 512];   // 16 rows x (2*Kin<=512) bf16

    const int  tid   = threadIdx.x;
    const int  K2    = Kin * 2;
    const long mbase = (long)blockIdx.x * 16;

    // ---- Phase 1: gather ------------------------------------------------
    {
        const int r    = tid >> 4;          // dst row 0..15
        const int c16  = tid & 15;          // column group
        const int cpt  = Kin >> 4;          // floats per thread: 8 or 16
        const int nv   = cpt >> 2;          // float4 chunks: 2 or 4
        const int col0 = c16 * cpt;

        // self row
        long g = mbase + r;
        if (perm) g = perm[g];
        {
            const float4* s4 = (const float4*)(in + g * (long)Kin + col0);
            #pragma unroll
            for (int i = 0; i < 4; ++i) {
                if (i >= nv) break;
                float4 t4 = s4[i];
                Alds[r * K2 + col0 + 4 * i + 0] = f2bf(relu_maybe(t4.x, relu_in));
                Alds[r * K2 + col0 + 4 * i + 1] = f2bf(relu_maybe(t4.y, relu_in));
                Alds[r * K2 + col0 + 4 * i + 2] = f2bf(relu_maybe(t4.z, relu_in));
                Alds[r * K2 + col0 + 4 * i + 3] = f2bf(relu_maybe(t4.w, relu_in));
            }
        }

        // neighbor mean (relu applied per element BEFORE mean, as in reference)
        float4 acc[4];
        #pragma unroll
        for (int i = 0; i < 4; ++i) acc[i] = make_float4(0.f, 0.f, 0.f, 0.f);
        for (int f = 0; f < fan; ++f) {
            long nid = idx[(mbase + r) * fan + f];
            if (perm) nid = perm[nid];
            const float4* s4 = (const float4*)(in + nid * (long)Kin + col0);
            #pragma unroll
            for (int i = 0; i < 4; ++i) {
                if (i >= nv) break;
                float4 t4 = s4[i];
                acc[i].x += relu_maybe(t4.x, relu_in);
                acc[i].y += relu_maybe(t4.y, relu_in);
                acc[i].z += relu_maybe(t4.z, relu_in);
                acc[i].w += relu_maybe(t4.w, relu_in);
            }
        }
        const float sc = 1.f / (float)fan;
        #pragma unroll
        for (int i = 0; i < 4; ++i) {
            if (i >= nv) break;
            Alds[r * K2 + Kin + col0 + 4 * i + 0] = f2bf(acc[i].x * sc);
            Alds[r * K2 + Kin + col0 + 4 * i + 1] = f2bf(acc[i].y * sc);
            Alds[r * K2 + Kin + col0 + 4 * i + 2] = f2bf(acc[i].z * sc);
            Alds[r * K2 + Kin + col0 + 4 * i + 3] = f2bf(acc[i].w * sc);
        }
    }
    __syncthreads();

    // ---- Phase 2: WMMA GEMM --------------------------------------------
    const int lane = tid & 31;
    const int wave = tid >> 5;
    const int M    = lane & 15;
    const int kb8  = (lane >> 4) * 16;      // A K-base: lanes 16-31 start at K=16
    const int NKT  = K2 >> 5;               // K'-tiles of 32
    const int NTP  = Npad >> 4;             // N-tiles of 16
    const int ncol = lane & 15;
    const int rb   = (lane >> 4) * 8;       // C/D row base for this half-wave

    for (int nt = wave; nt < NTP; nt += 8) {
        v8f acc = {};
        for (int kt = 0; kt < NKT; ++kt) {
            const int ab = kt * 32 + kb8;
            union { uint4 u[2]; v16bf v; } A, B;
            A.u[0] = *(const uint4*)&Alds[M * K2 + ab];        // K = ab .. ab+7
            A.u[1] = *(const uint4*)&Alds[M * K2 + ab + 8];    // K = ab+8 .. ab+15
            const uint4* bp = (const uint4*)(Wp + (((size_t)nt * NKT + kt) * 32 + lane) * 16);
            B.u[0] = bp[0];
            B.u[1] = bp[1];
            acc = __builtin_amdgcn_wmma_f32_16x16x32_bf16(
                false, A.v, false, B.v, (short)0, acc, false, false);
        }
        const int n  = nt * 16 + ncol;
        const float bv = (n < Nout) ? bias[n] : 0.f;
        #pragma unroll
        for (int i = 0; i < 8; ++i) {
            float v = acc[i] + bv;
            if (relu_out) v = fmaxf(v, 0.f);
            if (n < Nout)
                out[(mbase + rb + i) * (long)Nout + n] = v;
        }
    }
}

// ---------------------------------------------------------------------------
extern "C" void kernel_launch(void* const* d_in, const int* in_sizes, int n_in,
                              void* d_out, int out_size, void* d_ws, size_t ws_size,
                              hipStream_t stream) {
    (void)in_sizes; (void)n_in; (void)out_size; (void)ws_size;

    const float* feats  = (const float*)d_in[0];
    const float* Ws0    = (const float*)d_in[1];
    const float* Wn0    = (const float*)d_in[2];
    const float* b0     = (const float*)d_in[3];
    const float* Ws1    = (const float*)d_in[4];
    const float* Wn1    = (const float*)d_in[5];
    const float* b1     = (const float*)d_in[6];
    const float* Ws2    = (const float*)d_in[7];
    const float* Wn2    = (const float*)d_in[8];
    const float* b2     = (const float*)d_in[9];
    const int*   idx0   = (const int*)d_in[10];
    const int*   idx1   = (const int*)d_in[11];
    const int*   idx2   = (const int*)d_in[12];
    const int*   invidx = (const int*)d_in[13];

    char* ws = (char*)d_ws;
    float*          h0  = (float*)(ws);                          // 51200*256*4
    float*          h1  = (float*)(ws + 52428800);               //  5120*256*4
    unsigned short* w0p = (unsigned short*)(ws + 57671680);      // 16*8*512*2
    unsigned short* w1p = (unsigned short*)(ws + 57802752);      // 16*16*512*2
    unsigned short* w2p = (unsigned short*)(ws + 58064896);      //  3*16*512*2

    // pack weights (bf16, WMMA B-fragment order)
    sage_pack_w<<<(16 * 8  * 512 + 255) / 256, 256, 0, stream>>>(Ws0, Wn0, w0p, 128, 256,  8, 16);
    sage_pack_w<<<(16 * 16 * 512 + 255) / 256, 256, 0, stream>>>(Ws1, Wn1, w1p, 256, 256, 16, 16);
    sage_pack_w<<<( 3 * 16 * 512 + 255) / 256, 256, 0, stream>>>(Ws2, Wn2, w2p, 256,  47, 16,  3);

    // Layer 0: feats -> h0 (raw; permute+relu folded into layer 1 loads)
    sage_layer<<<51200 / 16, 256, 0, stream>>>(
        feats, nullptr, idx0, w0p, b0, h0,
        /*Kin=*/128, /*fan=*/15, /*Nout=*/256, /*Npad=*/256, /*relu_in=*/0, /*relu_out=*/0);

    // Layer 1: h0 (via inverse_idx, relu-on-load) -> h1 (relu on store)
    sage_layer<<<5120 / 16, 256, 0, stream>>>(
        h0, invidx, idx1, w1p, b1, h1,
        /*Kin=*/256, /*fan=*/10, /*Nout=*/256, /*Npad=*/256, /*relu_in=*/1, /*relu_out=*/1);

    // Layer 2: h1 -> d_out (1024 x 47)
    sage_layer<<<1024 / 16, 256, 0, stream>>>(
        h1, nullptr, idx2, w2p, b2, (float*)d_out,
        /*Kin=*/256, /*fan=*/5, /*Nout=*/47, /*Npad=*/48, /*relu_in=*/0, /*relu_out=*/0);
}
